// HeteroGAT_71107478552873
// MI455X (gfx1250) — matrix operand order repaired and checked by reading, not compiled, not run on hardware
//
#include <hip/hip_runtime.h>
#include <hip/hip_bf16.h>

typedef __attribute__((ext_vector_type(16))) _Float16 v16h;
typedef __attribute__((ext_vector_type(8)))  float    v8f;

#define NU_    200000
#define NI_    100000
#define EE_    500000
#define NPAIR_ 100000
// padded row counts (multiples of 128) for guard-free WMMA store epilogues
#define NUP_   200064
#define NIP_   100096
#define NPP_   100096

// ---------- helpers ----------

__device__ __forceinline__ unsigned fenc(float f) {
    unsigned u = __float_as_uint(f);
    return (u & 0x80000000u) ? ~u : (u | 0x80000000u);
}
__device__ __forceinline__ float fdec(unsigned u) {
    return __uint_as_float((u & 0x80000000u) ? (u & 0x7FFFFFFFu) : ~u);
}

// Load one 16x32 f16 A-fragment slice for this lane.
// xr = row_base + kt*32 + (lane>>4)*8 ; per CDNA5 16-bit A layout the lane's
// 16 halves are k = base+{0..7} and k = base+16+{0..7}.
__device__ __forceinline__ v16h hg_load_a(const float* __restrict__ xr) {
    float4 f0 = *(const float4*)(xr);
    float4 f1 = *(const float4*)(xr + 4);
    float4 f2 = *(const float4*)(xr + 16);
    float4 f3 = *(const float4*)(xr + 20);
    v16h a;
    a[0]  = (_Float16)f0.x; a[1]  = (_Float16)f0.y; a[2]  = (_Float16)f0.z; a[3]  = (_Float16)f0.w;
    a[4]  = (_Float16)f1.x; a[5]  = (_Float16)f1.y; a[6]  = (_Float16)f1.z; a[7]  = (_Float16)f1.w;
    a[8]  = (_Float16)f2.x; a[9]  = (_Float16)f2.y; a[10] = (_Float16)f2.z; a[11] = (_Float16)f2.w;
    a[12] = (_Float16)f3.x; a[13] = (_Float16)f3.y; a[14] = (_Float16)f3.z; a[15] = (_Float16)f3.w;
    return a;
}

// ---------- prep kernels ----------

// vectorized zero: n4 = element count / 4 (all zeroed sizes are multiples of 4)
__global__ void hg_zero4(uint4* __restrict__ p, long long n4) {
    long long i = (long long)blockIdx.x * blockDim.x + threadIdx.x;
    if (i < n4) p[i] = make_uint4(0u, 0u, 0u, 0u);
}

// Swizzle a (nkt*32) x 64 f32 weight matrix into per-lane WMMA B fragments.
// Fragment (kt,nt): 32 lanes x 16 halves, lane L covers n = nt*16 + L%16,
// k = kt*32 + (e/8)*16 + (L/16)*8 + e%8  (e = 0..15).
__global__ void hg_swizzleW(const float* __restrict__ W, _Float16* __restrict__ dst, int nkt) {
    int total = nkt * 4 * 32 * 16;
    int idx = blockIdx.x * blockDim.x + threadIdx.x;
    if (idx >= total) return;
    int e    = idx & 15;
    int lane = (idx >> 4) & 31;
    int frag = idx >> 9;
    int kt = frag >> 2, nt = frag & 3;
    int khalf = lane >> 4;
    int n = (nt << 4) + (lane & 15);
    int j = e >> 1, p = e & 1;
    int k = kt * 32 + ((e >> 3) << 4) + khalf * 8 + (e & 7);
    dst[frag * 512 + lane * 16 + e] = (_Float16)W[k * 64 + n];
}

// v[r] = W[r] @ a[r]   (6 blocks x 64 threads)
__global__ void hg_wa(const float* __restrict__ W, const float* __restrict__ a,
                      float* __restrict__ v) {
    int r = blockIdx.x, d = threadIdx.x;
    const float* Wr = W + r * 4096;
    const float* ar = a + r * 64;
    float s = 0.f;
    #pragma unroll
    for (int h = 0; h < 64; h++) s += Wr[d * 64 + h] * ar[h];
    v[r * 64 + d] = s;
}

// ---------- WMMA projection: hs = X @ W  (M x 64) @ (64 x 64), hs stored f16 ----------
// hs has rows padded to a multiple of 128: stores are guard-free.

__global__ void __launch_bounds__(256)
hg_proj(const float* __restrict__ X, int M,
        const _Float16* __restrict__ Wsw, _Float16* __restrict__ hs) {
    int lane = threadIdx.x & 31, wave = threadIdx.x >> 5;
    int row0 = blockIdx.x * 128 + wave * 16;
    int khalf = lane >> 4, mloc = lane & 15;
    int rowA = row0 + mloc;
    int rc = rowA < M ? rowA : (M - 1);   // clamp loads: keep EXEC all-1s for WMMA

    v16h a[2];
    #pragma unroll
    for (int kt = 0; kt < 2; kt++)
        a[kt] = hg_load_a(X + (long long)rc * 64 + kt * 32 + khalf * 8);

    #pragma unroll
    for (int nt = 0; nt < 4; nt++) {
        v8f acc = {};
        #pragma unroll
        for (int kt = 0; kt < 2; kt++) {
            v16h b = *(const v16h*)(Wsw + (kt * 4 + nt) * 512 + lane * 16);
            acc = __builtin_amdgcn_wmma_f32_16x16x32_f16(false, a[kt], false, b,
                                                         (short)0, acc, false, false);
        }
        int col = nt * 16 + (lane & 15);
        _Float16* base = hs + (long long)(row0 + khalf * 8) * 64 + col;
        #pragma unroll
        for (int vg = 0; vg < 8; vg++) base[vg * 64] = (_Float16)acc[vg];
    }
}

// al[row] = X[row,:] . v   (attention logit via folded vector)
__global__ void hg_matvec(const float* __restrict__ X, int M,
                          const float* __restrict__ v, float* __restrict__ al) {
    int r = blockIdx.x * blockDim.x + threadIdx.x;
    if (r >= M) return;
    const float* x = X + (long long)r * 64;
    float s = 0.f;
    #pragma unroll
    for (int k = 0; k < 64; k++) s += x[k] * v[k];
    al[r] = s;
}

// ---------- edge passes ----------

__global__ void hg_edge1(const int* __restrict__ src, const int* __restrict__ dst,
                         const float* __restrict__ als, const float* __restrict__ ald,
                         float* __restrict__ ebuf, unsigned* __restrict__ m) {
    int e = blockIdx.x * blockDim.x + threadIdx.x;
    if (e >= EE_) return;
    float v = als[src[e]] + ald[dst[e]];
    float ev = v > 0.f ? v : 0.2f * v;   // leaky_relu(0.2)
    ebuf[e] = ev;
    atomicMax(&m[dst[e]], fenc(ev));
}

__global__ void hg_edge2(const int* __restrict__ dst, float* __restrict__ ebuf,
                         const unsigned* __restrict__ m, float* __restrict__ den) {
    int e = blockIdx.x * blockDim.x + threadIdx.x;
    if (e >= EE_) return;
    int d = dst[e];
    float ex = expf(ebuf[e] - fdec(m[d]));
    ebuf[e] = ex;
    atomicAdd(&den[d], ex);
}

// scatter: out[dst] += alpha * hs[src]   (32 lanes per edge, 2 cols per lane)
__global__ void hg_edge3(const int* __restrict__ src, const int* __restrict__ dst,
                         const float* __restrict__ ebuf, const float* __restrict__ den,
                         const _Float16* __restrict__ hs, float* __restrict__ out) {
    long long t = (long long)blockIdx.x * blockDim.x + threadIdx.x;
    int e = (int)(t >> 5);
    int lane = (int)(t & 31);
    if (e >= EE_) return;
    int s = src[e], d = dst[e];
    float alpha = ebuf[e] / den[d];
    unsigned hu = *(const unsigned*)(hs + (long long)s * 64 + lane * 2);  // 2 packed halves
    const _Float16* hp = (const _Float16*)&hu;
    float* o = out + (long long)d * 64 + lane * 2;
    atomicAdd(o + 0, alpha * (float)hp[0]);
    atomicAdd(o + 1, alpha * (float)hp[1]);
}

// x += b[r0]+b[r0+1]+b[r0+2]; optional relu
__global__ void hg_biasrelu(float* __restrict__ x, int M,
                            const float* __restrict__ b3, int dorelu) {
    long long i = (long long)blockIdx.x * blockDim.x + threadIdx.x;
    if (i >= (long long)M * 64) return;
    int h = (int)(i & 63);
    float v = x[i] + b3[h] + b3[64 + h] + b3[128 + h];
    if (dorelu) v = v > 0.f ? v : 0.f;
    x[i] = v;
}

// ---------- decoder ----------

// h = relu(concat(xu[uid], xi[iid]) @ Wd1 + bd1), WMMA over K=128.
// hout rows padded to NPP_: stores are guard-free.
__global__ void __launch_bounds__(256)
hg_dec1(const float* __restrict__ xu, const float* __restrict__ xi,
        const int* __restrict__ uid, const int* __restrict__ iid,
        const _Float16* __restrict__ Wsw, const float* __restrict__ bd1,
        float* __restrict__ hout) {
    int lane = threadIdx.x & 31, wave = threadIdx.x >> 5;
    int p0 = blockIdx.x * 128 + wave * 16;
    int khalf = lane >> 4, mloc = lane & 15;
    int p = p0 + mloc;
    int pc = p < NPAIR_ ? p : (NPAIR_ - 1);
    int u = uid[pc], it = iid[pc];

    v16h a[4];
    #pragma unroll
    for (int kt = 0; kt < 4; kt++) {
        const float* base = (kt < 2) ? (xu + (long long)u * 64 + kt * 32)
                                     : (xi + (long long)it * 64 + (kt - 2) * 32);
        a[kt] = hg_load_a(base + khalf * 8);
    }

    #pragma unroll
    for (int nt = 0; nt < 4; nt++) {
        v8f acc = {};
        #pragma unroll
        for (int kt = 0; kt < 4; kt++) {
            v16h b = *(const v16h*)(Wsw + (kt * 4 + nt) * 512 + lane * 16);
            acc = __builtin_amdgcn_wmma_f32_16x16x32_f16(false, a[kt], false, b,
                                                         (short)0, acc, false, false);
        }
        int col = nt * 16 + (lane & 15);
        float bb = bd1[col];
        float* base = hout + (long long)(p0 + khalf * 8) * 64 + col;
        #pragma unroll
        for (int vg = 0; vg < 8; vg++) {
            float hv = acc[vg] + bb;
            base[vg * 64] = hv > 0.f ? hv : 0.f;
        }
    }
}

// out[p, 0..3] = h[p,:] @ Wd2 + bd2   (one thread per pair; row read once)
__global__ void hg_dec2(const float* __restrict__ h, const float* __restrict__ Wd2,
                        const float* __restrict__ bd2, float* __restrict__ out) {
    int p = blockIdx.x * blockDim.x + threadIdx.x;
    if (p >= NPAIR_) return;
    const float* hp = h + (long long)p * 64;
    float s0 = bd2[0], s1 = bd2[1], s2 = bd2[2], s3 = bd2[3];
    #pragma unroll
    for (int k = 0; k < 64; k++) {
        float hv = hp[k];
        s0 += hv * Wd2[k * 4 + 0];
        s1 += hv * Wd2[k * 4 + 1];
        s2 += hv * Wd2[k * 4 + 2];
        s3 += hv * Wd2[k * 4 + 3];
    }
    float4* o = (float4*)(out + (long long)p * 4);
    *o = make_float4(s0, s1, s2, s3);
}

// ---------- host launch ----------

extern "C" void kernel_launch(void* const* d_in, const int* in_sizes, int n_in,
                              void* d_out, int out_size, void* d_ws, size_t ws_size,
                              hipStream_t stream) {
    const float* user_emb = (const float*)d_in[0];
    const float* item_emb = (const float*)d_in[1];
    const float* W1_src   = (const float*)d_in[2];
    const float* W1_dst   = (const float*)d_in[3];
    const float* a1_src   = (const float*)d_in[4];
    const float* a1_dst   = (const float*)d_in[5];
    const float* b1       = (const float*)d_in[6];
    const float* W2_src   = (const float*)d_in[7];
    const float* W2_dst   = (const float*)d_in[8];
    const float* a2_src   = (const float*)d_in[9];
    const float* a2_dst   = (const float*)d_in[10];
    const float* b2       = (const float*)d_in[11];
    const float* Wd1      = (const float*)d_in[12];
    const float* bd1      = (const float*)d_in[13];
    const float* Wd2      = (const float*)d_in[14];
    const float* bd2      = (const float*)d_in[15];
    const int*   edges    = (const int*)d_in[16];
    const int*   user_ids = (const int*)d_in[17];
    const int*   item_ids = (const int*)d_in[18];
    float* out = (float*)d_out;

    // workspace carve-up
    char* ws = (char*)d_ws;
    size_t off = 0;
    auto carve = [&](size_t bytes) -> char* {
        char* p = ws + off;
        off = (off + bytes + 255) & ~(size_t)255;
        return p;
    };
    _Float16* hsBuf = (_Float16*)carve(sizeof(_Float16) * 64ull * (3ull * NUP_ + 3ull * NIP_));
    _Float16* wsw1  = (_Float16*)carve(sizeof(_Float16) * 6 * 4096);
    _Float16* wsw2  = (_Float16*)carve(sizeof(_Float16) * 6 * 4096);
    _Float16* wswd  = (_Float16*)carve(sizeof(_Float16) * 8192);
    float* x1u  = (float*)carve(4ull * NU_ * 64);
    float* x1i  = (float*)carve(4ull * NI_ * 64);
    float* x2u  = (float*)carve(4ull * NU_ * 64);
    float* x2i  = (float*)carve(4ull * NI_ * 64);
    float* alS  = (float*)carve(4ull * (3ull * NU_ + 3ull * NI_));
    float* alD  = (float*)carve(4ull * (3ull * NI_ + 3ull * NU_));
    unsigned* mEnc = (unsigned*)carve(4ull * NU_);
    float* den  = (float*)carve(4ull * NU_);
    float* ebuf = (float*)carve(4ull * EE_);
    float* dech = (float*)carve(4ull * NPP_ * 64);
    float* vs1  = (float*)carve(4ull * 6 * 64);
    float* vd1  = (float*)carve(4ull * 6 * 64);
    float* vs2  = (float*)carve(4ull * 6 * 64);
    float* vd2  = (float*)carve(4ull * 6 * 64);

    // hs regions use padded row counts so hg_proj can store guard-free
    const size_t hsOff[6] = {0, (size_t)NUP_ * 64, 2ull * NUP_ * 64, 3ull * NUP_ * 64,
                             3ull * NUP_ * 64 + (size_t)NIP_ * 64,
                             3ull * NUP_ * 64 + 2ull * NIP_ * 64};
    const size_t aSOff[6] = {0, NU_, 2ull * NU_, 3ull * NU_, 3ull * NU_ + NI_, 3ull * NU_ + 2ull * NI_};
    const size_t aDOff[6] = {0, NI_, 2ull * NI_, 3ull * NI_, 3ull * NI_ + NU_, 3ull * NI_ + 2ull * NU_};

    // --- prep: weight swizzles + folded attention vectors
    for (int r = 0; r < 6; r++) {
        hg_swizzleW<<<(4096 + 255) / 256, 256, 0, stream>>>(W1_src + r * 4096, wsw1 + r * 4096, 2);
        hg_swizzleW<<<(4096 + 255) / 256, 256, 0, stream>>>(W2_src + r * 4096, wsw2 + r * 4096, 2);
    }
    hg_swizzleW<<<(8192 + 255) / 256, 256, 0, stream>>>(Wd1, wswd, 4);
    hg_wa<<<6, 64, 0, stream>>>(W1_src, a1_src, vs1);
    hg_wa<<<6, 64, 0, stream>>>(W1_dst, a1_dst, vd1);
    hg_wa<<<6, 64, 0, stream>>>(W2_src, a2_src, vs2);
    hg_wa<<<6, 64, 0, stream>>>(W2_dst, a2_dst, vd2);

    for (int L = 0; L < 2; L++) {
        const float* Xu = (L == 0) ? user_emb : x1u;
        const float* Xi = (L == 0) ? item_emb : x1i;
        float* outU = (L == 0) ? x1u : x2u;
        float* outI = (L == 0) ? x1i : x2i;
        const _Float16* wswL = (L == 0) ? wsw1 : wsw2;
        const float* vsL = (L == 0) ? vs1 : vs2;
        const float* vdL = (L == 0) ? vd1 : vd2;
        const float* bL  = (L == 0) ? b1 : b2;

        long long nUe = (long long)NU_ * 64, nIe = (long long)NI_ * 64;
        hg_zero4<<<(unsigned)((nUe / 4 + 255) / 256), 256, 0, stream>>>((uint4*)outU, nUe / 4);
        hg_zero4<<<(unsigned)((nIe / 4 + 255) / 256), 256, 0, stream>>>((uint4*)outI, nIe / 4);

        for (int r = 0; r < 6; r++) {
            const float* Xs = (r < 3) ? Xu : Xi;  int Ms = (r < 3) ? NU_ : NI_;
            const float* Xd = (r < 3) ? Xi : Xu;  int Md = (r < 3) ? NI_ : NU_;
            hg_proj<<<(Ms + 127) / 128, 256, 0, stream>>>(Xs, Ms, wswL + r * 4096, hsBuf + hsOff[r]);
            hg_matvec<<<(Ms + 255) / 256, 256, 0, stream>>>(Xs, Ms, vsL + r * 64, alS + aSOff[r]);
            hg_matvec<<<(Md + 255) / 256, 256, 0, stream>>>(Xd, Md, vdL + r * 64, alD + aDOff[r]);
        }
        for (int r = 0; r < 6; r++) {
            int Md = (r < 3) ? NI_ : NU_;
            float* dstOut = (r < 3) ? outI : outU;
            const int* src = edges + (size_t)r * 2 * EE_;
            const int* dst = src + EE_;
            hg_zero4<<<(Md / 4 + 255) / 256, 256, 0, stream>>>((uint4*)mEnc, Md / 4);
            hg_zero4<<<(Md / 4 + 255) / 256, 256, 0, stream>>>((uint4*)den, Md / 4);
            hg_edge1<<<(EE_ + 255) / 256, 256, 0, stream>>>(src, dst, alS + aSOff[r], alD + aDOff[r], ebuf, mEnc);
            hg_edge2<<<(EE_ + 255) / 256, 256, 0, stream>>>(dst, ebuf, mEnc, den);
            long long nt3 = (long long)EE_ * 32;
            hg_edge3<<<(unsigned)((nt3 + 255) / 256), 256, 0, stream>>>(src, dst, ebuf, den, hsBuf + hsOff[r], dstOut);
        }
        hg_biasrelu<<<(unsigned)((nIe + 255) / 256), 256, 0, stream>>>(outI, NI_, bL, L == 0 ? 1 : 0);
        hg_biasrelu<<<(unsigned)((nUe + 255) / 256), 256, 0, stream>>>(outU, NU_, bL + 3 * 64, L == 0 ? 1 : 0);
    }

    // decoder
    hg_dec1<<<(NPAIR_ + 127) / 128, 256, 0, stream>>>(x2u, x2i, user_ids, item_ids, wswd, bd1, dech);
    hg_dec2<<<(NPAIR_ + 255) / 256, 256, 0, stream>>>(dech, Wd2, bd2, out);
}
